// Embeddings_62208306315589
// MI455X (gfx1250) — matrix operand order repaired and checked by reading, not compiled, not run on hardware
//
#include <hip/hip_runtime.h>
#include <cstdint>
#include <cstddef>

// ---------------------------------------------------------------------------
// Problem constants (from reference)
// ---------------------------------------------------------------------------
#define FEA   256
#define ATT   64
#define HLAST 10
#define STEPS 12
#define HID   2560          // LSTM hidden size
#define INSZ  5120          // 2*HID (concat input)
#define BATCH 64
#define NTILES 160          // HID / 16  (N tiles per gate)
#define KCHUNKS 160         // INSZ / 32 (K chunks of 32)
#define RING  12            // LDS pipeline depth (slots of A+B chunk tiles)

typedef __bf16 bf16_t;
typedef __attribute__((ext_vector_type(16))) __bf16 bf16x16;
typedef __attribute__((ext_vector_type(8)))  float  f32x8;
typedef __attribute__((ext_vector_type(4)))  int    v4i_t;

union FragAB { uint4 u[2]; bf16x16 v; };

// ---------------------------------------------------------------------------
// Async global->LDS copy (gfx1250). Guarded: falls back to load+ds_store.
// Builtin signature (per hipcc diagnostic): (v4i AS1*, v4i AS3*, imm, imm)
// ---------------------------------------------------------------------------
#if __has_builtin(__builtin_amdgcn_global_load_async_to_lds_b128)
#define HAS_ASYNC_LDS 1
#else
#define HAS_ASYNC_LDS 0
#endif

__device__ __forceinline__ void cp16_g2l(const void* g, void* l) {
#if HAS_ASYNC_LDS
  __builtin_amdgcn_global_load_async_to_lds_b128(
      (__attribute__((address_space(1))) v4i_t*)g,
      (__attribute__((address_space(3))) v4i_t*)l,
      0, 0);
#else
  *(uint4*)l = *(const uint4*)g;
#endif
}

template <int N>
__device__ __forceinline__ void wait_async() {
#if HAS_ASYNC_LDS
#if __has_builtin(__builtin_amdgcn_s_wait_asynccnt)
  __builtin_amdgcn_s_wait_asynccnt(N);
#else
  asm volatile("s_wait_asynccnt %0" ::"i"(N) : "memory");
#endif
#endif
}

// ---------------------------------------------------------------------------
// Kernel 1: one-time weight pack  fp32 [K=5120, N=2560] row-major
//   -> bf16 B-fragment tiles: [gate][ntile][kchunk][lane*32B]
//   B order used here (contiguous k, not ISA table): lanes 0-15: N=lane,  K=2v..2v+1
//                             lanes16-31: N=lane-16, K=16+2v..17+2v
// ---------------------------------------------------------------------------
__global__ __launch_bounds__(128)
void pack_weights(const float* __restrict__ Wf, const float* __restrict__ Wi,
                  const float* __restrict__ Wo, const float* __restrict__ Wc,
                  bf16_t* __restrict__ packed) {
  size_t tile = (size_t)blockIdx.x * 4 + (threadIdx.x >> 5);
  int lane = threadIdx.x & 31;
  int g    = (int)(tile / (NTILES * KCHUNKS));
  int rem  = (int)(tile % (NTILES * KCHUNKS));
  int nt   = rem / KCHUNKS;
  int c    = rem % KCHUNKS;
  const float* W = (g == 0) ? Wf : (g == 1) ? Wi : (g == 2) ? Wo : Wc;
  int n  = nt * 16 + (lane & 15);
  int kb = c * 32 + (lane >> 4) * 16;
  bf16_t* out = packed + tile * 512 + (size_t)lane * 16;   // 512 bf16 per tile
#pragma unroll
  for (int v = 0; v < 8; ++v) {
    int k = kb + v * 2;
    out[v * 2 + 0] = (bf16_t)W[(size_t)(k + 0) * HID + n];
    out[v * 2 + 1] = (bf16_t)W[(size_t)(k + 1) * HID + n];
  }
}

// ---------------------------------------------------------------------------
// Kernel 2: fused encoder  relu(relu(x@W0+b0)@W1+b1) -> bf16 xseq [B][T][HID]
// one block per (b*T+t) sample, one thread per FEA row
// ---------------------------------------------------------------------------
__global__ __launch_bounds__(256)
void encoder_fused(const float* __restrict__ x, const float* __restrict__ W0,
                   const float* __restrict__ b0, const float* __restrict__ W1,
                   const float* __restrict__ b1, bf16_t* __restrict__ xseq) {
  __shared__ float sW0[ATT * 128];
  __shared__ float sW1[128 * HLAST];
  __shared__ float sb0[128];
  __shared__ float sb1[HLAST];
  int tid = threadIdx.x;
  for (int i = tid; i < ATT * 128; i += 256) sW0[i] = W0[i];
  for (int i = tid; i < 128 * HLAST; i += 256) sW1[i] = W1[i];
  if (tid < 128) sb0[tid] = b0[tid];
  if (tid < HLAST) sb1[tid] = b1[tid];
  __syncthreads();

  int bt = blockIdx.x;
  const float* xp = x + ((size_t)bt * FEA + tid) * ATT;
  float xr[ATT];
#pragma unroll
  for (int j = 0; j < ATT; ++j) xr[j] = xp[j];

  float outv[HLAST];
#pragma unroll
  for (int l = 0; l < HLAST; ++l) outv[l] = sb1[l];

  for (int o = 0; o < 128; ++o) {
    float a = sb0[o];
#pragma unroll
    for (int j = 0; j < ATT; ++j) a = fmaf(xr[j], sW0[j * 128 + o], a);
    a = fmaxf(a, 0.0f);
#pragma unroll
    for (int l = 0; l < HLAST; ++l) outv[l] = fmaf(a, sW1[o * HLAST + l], outv[l]);
  }
  bf16_t* op = xseq + (size_t)bt * HID + (size_t)tid * HLAST;
#pragma unroll
  for (int l = 0; l < HLAST; ++l) op[l] = (bf16_t)fmaxf(outv[l], 0.0f);
}

// ---------------------------------------------------------------------------
// Kernel 3: zero LSTM state
// ---------------------------------------------------------------------------
__global__ void lstm_init(float* __restrict__ cst, bf16_t* __restrict__ hA,
                          bf16_t* __restrict__ hB) {
  int i = blockIdx.x * blockDim.x + threadIdx.x;
  if (i < BATCH * HID) {
    cst[i] = 0.0f;
    hA[i] = (bf16_t)0.0f;
    hB[i] = (bf16_t)0.0f;
  }
}

// ---------------------------------------------------------------------------
// Kernel 4: one LSTM step, fully fused (4 gate GEMM slices + cell update).
// grid = 160 WGs (one 16-col slice of HID), block = 128 (4 waves = 4 gates).
// Both A (comb activations) and B (packed bf16 weights) are streamed through
// a RING-deep LDS pipeline with async global->LDS copies, so the single wave
// per SIMD has ~RING chunks of latency tolerance (covers L2/HBM round trips).
// Per chunk/wave: 10x ds_load_b128 then 4 back-to-back v_wmma.
// ---------------------------------------------------------------------------
__global__ __launch_bounds__(128, 1)
void lstm_step_wmma(const bf16_t* __restrict__ packedW,
                    const bf16_t* __restrict__ xseq,
                    const bf16_t* __restrict__ hbf_in,
                    bf16_t* __restrict__ hbf_out,
                    float* __restrict__ c_state,
                    float* __restrict__ h_out,
                    const float* __restrict__ bfv, const float* __restrict__ biv,
                    const float* __restrict__ bov, const float* __restrict__ bcv,
                    int t) {
  // slot layout: [0,4096) A tile 64x32 bf16 ; [4096,8192) B tiles (4 gates x 1KB)
  __shared__ unsigned char sTile[RING * 8192];
  __shared__ float sGate[4][BATCH * 16];       // gate pre-activations

  const int tid  = threadIdx.x;
  const int wave = tid >> 5;                   // 0..3 -> gate f,i,o,c
  const int lane = tid & 31;
  const int half = lane >> 4;
  const int ln   = lane & 15;
  const int nt   = blockIdx.x;

  // stage chunk c into ring slot c%RING: 4 async b128 per thread (2 A + 2 B).
  // A: k < 2560 comes from xseq[:, t, :]; k >= 2560 from previous hidden.
  auto stage = [&](int c) {
    unsigned char* slot = sTile + (size_t)(c % RING) * 8192;
    const unsigned char* gsrc;
    size_t rstride;
    if (c < 80) {
      gsrc = (const unsigned char*)xseq + ((size_t)t * HID + (size_t)c * 32) * 2;
      rstride = (size_t)STEPS * HID * 2;
    } else {
      gsrc = (const unsigned char*)hbf_in + ((size_t)(c - 80) * 32) * 2;
      rstride = (size_t)HID * 2;
    }
#pragma unroll
    for (int s = 0; s < 2; ++s) {
      int f = tid * 16 + s * 2048;             // byte offset into 4KB A tile
      int row = f >> 6, col = f & 63;          // 64B per row (32 bf16)
      cp16_g2l(gsrc + (size_t)row * rstride + col, slot + f);
    }
    // B: each thread stages its own wave's gate tile fragment (32B, packed)
    const unsigned char* bsrc =
        (const unsigned char*)packedW +
        (((size_t)wave * NTILES + nt) * KCHUNKS + c) * 1024 + (size_t)lane * 32;
    unsigned char* bdst = slot + 4096 + wave * 1024 + lane * 32;
    cp16_g2l(bsrc, bdst);
    cp16_g2l(bsrc + 16, bdst + 16);
  };

  f32x8 acc[4];
#pragma unroll
  for (int m = 0; m < 4; ++m)
    acc[m] = (f32x8){0.f, 0.f, 0.f, 0.f, 0.f, 0.f, 0.f, 0.f};

  // prologue: fill RING-1 slots
#pragma unroll 1
  for (int p = 0; p < RING - 1; ++p) stage(p);

  for (int c = 0; c < KCHUNKS; ++c) {
    if (c + RING - 1 < KCHUNKS) {
      stage(c + RING - 1);                     // writes slot (c-1)%RING (free)
      wait_async<4 * (RING - 1)>();            // stage c complete (in-order)
    } else {
      wait_async<0>();                         // tail drain
    }
    __syncthreads();                           // slot c visible to all waves

    unsigned char* slot = sTile + (size_t)(c % RING) * 8192;

    FragAB fb;
    {
      const uint4* pb = (const uint4*)(slot + 4096 + wave * 1024 + lane * 32);
      fb.u[0] = pb[0];
      fb.u[1] = pb[1];
    }
    FragAB fa[4];
#pragma unroll
    for (int m = 0; m < 4; ++m) {
      // A fragment, 16-bit 16x32 layout: lane -> row m*16+ln,
      // two b128 at row*64 + half*32 and +16.
      const uint4* pa = (const uint4*)(slot + (m * 16 + ln) * 64 + half * 32);
      fa[m].u[0] = pa[0];
      fa[m].u[1] = pa[1];
    }
#pragma unroll
    for (int m = 0; m < 4; ++m)
      acc[m] = __builtin_amdgcn_wmma_f32_16x16x32_bf16(
          false, fa[m].v, false, fb.v, (short)0, acc[m], false, false);

    __syncthreads();                           // done reading slot before reuse
  }

  // scatter pre-activations: C/D layout lanes0-15 M=r, lanes16-31 M=8+r
#pragma unroll
  for (int m = 0; m < 4; ++m)
#pragma unroll
    for (int r = 0; r < 8; ++r)
      sGate[wave][(m * 16 + half * 8 + r) * 16 + ln] = acc[m][r];
  __syncthreads();

  // fused LSTM cell update: 1024 elements / 128 threads
#pragma unroll
  for (int e = 0; e < 8; ++e) {
    int idx = tid + e * 128;
    int row = idx >> 4;
    int n   = nt * 16 + (idx & 15);
    float pf = sGate[0][idx] + bfv[n];
    float pi = sGate[1][idx] + biv[n];
    float po = sGate[2][idx] + bov[n];
    float pg = sGate[3][idx] + bcv[n];
    float fg = 1.0f / (1.0f + __expf(-pf));
    float ig = 1.0f / (1.0f + __expf(-pi));
    float og = 1.0f / (1.0f + __expf(-po));
    float gg = tanhf(pg);
    size_t off = (size_t)row * HID + n;
    float cnew = fg * c_state[off] + ig * gg;
    float hnew = og * tanhf(cnew);
    c_state[off] = cnew;
    h_out[off]   = hnew;
    hbf_out[off] = (bf16_t)hnew;
  }
}

// ---------------------------------------------------------------------------
// Kernel 5: out = relu(hs).reshape(B,256,10) @ Wfc + bfc
// ---------------------------------------------------------------------------
__global__ __launch_bounds__(256)
void final_fc(const float* __restrict__ hs, const float* __restrict__ Wfc,
              const float* __restrict__ bfc, float* __restrict__ out) {
  __shared__ float sW[HLAST * HLAST];
  __shared__ float sb[HLAST];
  if (threadIdx.x < HLAST * HLAST) sW[threadIdx.x] = Wfc[threadIdx.x];
  if (threadIdx.x < HLAST) sb[threadIdx.x] = bfc[threadIdx.x];
  __syncthreads();
  int b = blockIdx.x, fea = threadIdx.x;
  const float* hp = hs + (size_t)b * HID + (size_t)fea * HLAST;
  float o[HLAST];
#pragma unroll
  for (int l = 0; l < HLAST; ++l) o[l] = sb[l];
#pragma unroll
  for (int j = 0; j < HLAST; ++j) {
    float v = fmaxf(hp[j], 0.0f);
#pragma unroll
    for (int l = 0; l < HLAST; ++l) o[l] = fmaf(v, sW[j * HLAST + l], o[l]);
  }
  float* op = out + ((size_t)b * FEA + fea) * HLAST;
#pragma unroll
  for (int l = 0; l < HLAST; ++l) op[l] = o[l];
}

// ---------------------------------------------------------------------------
// Host launcher
// ---------------------------------------------------------------------------
extern "C" void kernel_launch(void* const* d_in, const int* in_sizes, int n_in,
                              void* d_out, int out_size, void* d_ws,
                              size_t ws_size, hipStream_t stream) {
  const float* x   = (const float*)d_in[0];
  const float* W0  = (const float*)d_in[1];
  const float* b0  = (const float*)d_in[2];
  const float* W1  = (const float*)d_in[3];
  const float* b1  = (const float*)d_in[4];
  const float* Wf  = (const float*)d_in[5];
  const float* bfv = (const float*)d_in[6];
  const float* Wi  = (const float*)d_in[7];
  const float* biv = (const float*)d_in[8];
  const float* Wo  = (const float*)d_in[9];
  const float* bov = (const float*)d_in[10];
  const float* Wc  = (const float*)d_in[11];
  const float* bcv = (const float*)d_in[12];
  const float* Wfc = (const float*)d_in[13];
  const float* bfc = (const float*)d_in[14];
  float* out = (float*)d_out;

  char* ws = (char*)d_ws;
  auto alignup = [](size_t v) { return (v + 255) & ~(size_t)255; };
  size_t off = 0;
  bf16_t* packed = (bf16_t*)(ws + off);
  off = alignup(off + (size_t)4 * NTILES * KCHUNKS * 1024);   // 104.9 MB
  bf16_t* xseq = (bf16_t*)(ws + off);
  off = alignup(off + (size_t)BATCH * STEPS * HID * 2);       // 3.9 MB
  bf16_t* hbfA = (bf16_t*)(ws + off);
  off = alignup(off + (size_t)BATCH * HID * 2);
  bf16_t* hbfB = (bf16_t*)(ws + off);
  off = alignup(off + (size_t)BATCH * HID * 2);
  float* cst = (float*)(ws + off);
  off = alignup(off + (size_t)BATCH * HID * 4);
  float* h32 = (float*)(ws + off);
  off = alignup(off + (size_t)BATCH * HID * 4);
  (void)ws_size; (void)in_sizes; (void)n_in; (void)out_size;

  pack_weights<<<(4 * NTILES * KCHUNKS) / 4, 128, 0, stream>>>(Wf, Wi, Wo, Wc,
                                                               packed);
  encoder_fused<<<BATCH * STEPS, FEA, 0, stream>>>(x, W0, b0, W1, b1, xseq);
  lstm_init<<<(BATCH * HID + 255) / 256, 256, 0, stream>>>(cst, hbfA, hbfB);

  for (int t = 0; t < STEPS; ++t) {
    const bf16_t* hin = (t & 1) ? hbfB : hbfA;
    bf16_t* hout      = (t & 1) ? hbfA : hbfB;
    lstm_step_wmma<<<NTILES, 128, 0, stream>>>(packed, xseq, hin, hout, cst,
                                               h32, bfv, biv, bov, bcv, t);
  }

  final_fc<<<BATCH, FEA, 0, stream>>>(h32, Wfc, bfc, out);
}
